// NeuroPredessor_41549513621561
// MI455X (gfx1250) — matrix operand order repaired and checked by reading, not compiled, not run on hardware
//
#include <hip/hip_runtime.h>
#include <hip/hip_bf16.h>

#define N_VAR 4096
#define N_NODE 8192
#define N_CELLS 65536
#define DIM 128
#define N_ROUNDS 16

typedef __attribute__((ext_vector_type(16))) __bf16 v16bf;
typedef __attribute__((ext_vector_type(8))) float v8f;

// ---------------------------------------------------------------------------
// WMMA helpers (CDNA5 V_WMMA_F32_16X16X32_BF16, wave32)
// ---------------------------------------------------------------------------
__device__ __forceinline__ v8f wmma_bf16(v16bf a, v16bf b, v8f c) {
  // (neg_a, A, neg_b, B, c_mod, C, reuse_a, reuse_b)
  return __builtin_amdgcn_wmma_f32_16x16x32_bf16(false, a, false, b, (short)0, c,
                                                 false, false);
}

// A fragment (16x32, bf16) from a row-major f32 tile (16 rows x ld), converting
// f32 -> bf16 on load.  ISA layout: lanes 0-15 = rows 0-15 with K {0-7,16-23},
// lanes 16-31 = rows 0-15 with K {8-15,24-31}.
__device__ __forceinline__ v16bf a_frag_f32(const float* __restrict__ tile, int ld,
                                            int kbase, int lane) {
  const int lh = lane >> 4;
  const int m = lane & 15;
  const float* r = tile + (size_t)m * ld + kbase + lh * 8;
  v16bf a;
#pragma unroll
  for (int i = 0; i < 8; ++i) a[i] = (__bf16)r[i];
#pragma unroll
  for (int i = 0; i < 8; ++i) a[8 + i] = (__bf16)r[16 + i];
  return a;
}

// Same, but source tile already bf16 (LDS staging buffer).
__device__ __forceinline__ v16bf a_frag_bf16(const __bf16* __restrict__ tile, int ld,
                                             int kbase, int lane) {
  const int lh = lane >> 4;
  const int m = lane & 15;
  const __bf16* r = tile + (size_t)m * ld + kbase + lh * 8;
  v16bf a;
#pragma unroll
  for (int i = 0; i < 8; ++i) a[i] = r[i];
#pragma unroll
  for (int i = 0; i < 8; ++i) a[8 + i] = r[16 + i];
  return a;
}

// B fragment (32x16, bf16) implementing x @ W^T: B[k][n] = W[nbase+n][kbase+k],
// W row-major [*, DIM].  ISA layout: lanes 0-15 hold K {0-7,16-23} (col = lane&15),
// lanes 16-31 hold K {8-15,24-31} -> matches the A fragment K order.
__device__ __forceinline__ v16bf b_frag_wT(const __bf16* __restrict__ W, int nbase,
                                           int kbase, int lane) {
  const int lh = lane >> 4;
  const int n = nbase + (lane & 15);
  const __bf16* p = W + (size_t)n * DIM + kbase + lh * 8;
  v16bf b;
#pragma unroll
  for (int i = 0; i < 8; ++i) b[i] = p[i];
#pragma unroll
  for (int i = 0; i < 8; ++i) b[8 + i] = p[16 + i];
  return b;
}

__device__ __forceinline__ float sigmoidf(float v) {
  return 1.0f / (1.0f + __expf(-v));
}

// ---------------------------------------------------------------------------
// Fused 3-layer MLP: y = (relu(relu(x W1^T + b1) W2^T + b2)) W3^T + b3
// One wave per 16 rows; LDS staging between layers (wave-private, DS in-order).
// 96 WMMAs per wave.
// ---------------------------------------------------------------------------
__global__ __launch_bounds__(128) void mlp3_kernel(
    const float* __restrict__ x, float* __restrict__ y,
    const __bf16* __restrict__ W1, const float* __restrict__ b1,
    const __bf16* __restrict__ W2, const float* __restrict__ b2,
    const __bf16* __restrict__ W3, const float* __restrict__ b3) {
  __shared__ __bf16 stage[4][16 * DIM];
  const int lane = threadIdx.x & 31;
  const int wave = threadIdx.x >> 5;
  const int lh = lane >> 4;
  const int cl = lane & 15;
  const int mbase = blockIdx.x * 64 + wave * 16;
  __bf16* s = stage[wave];

  v16bf a[4];
#pragma unroll
  for (int k = 0; k < 4; ++k)
    a[k] = a_frag_f32(x + (size_t)mbase * DIM, DIM, k * 32, lane);

  // layer 1 -> relu -> LDS (bf16)
#pragma unroll
  for (int nt = 0; nt < 8; ++nt) {
    v8f acc = {};
#pragma unroll
    for (int k = 0; k < 4; ++k)
      acc = wmma_bf16(a[k], b_frag_wT(W1, nt * 16, k * 32, lane), acc);
    const float bias = b1[nt * 16 + cl];
#pragma unroll
    for (int v = 0; v < 8; ++v) {
      float t = acc[v] + bias;
      s[(v + 8 * lh) * DIM + nt * 16 + cl] = (__bf16)fmaxf(t, 0.0f);
    }
  }

  // layer 2 -> relu -> LDS (in place; all A loads precede stores, DS in-order)
#pragma unroll
  for (int k = 0; k < 4; ++k) a[k] = a_frag_bf16(s, DIM, k * 32, lane);
#pragma unroll
  for (int nt = 0; nt < 8; ++nt) {
    v8f acc = {};
#pragma unroll
    for (int k = 0; k < 4; ++k)
      acc = wmma_bf16(a[k], b_frag_wT(W2, nt * 16, k * 32, lane), acc);
    const float bias = b2[nt * 16 + cl];
#pragma unroll
    for (int v = 0; v < 8; ++v) {
      float t = acc[v] + bias;
      s[(v + 8 * lh) * DIM + nt * 16 + cl] = (__bf16)fmaxf(t, 0.0f);
    }
  }

  // layer 3 -> global f32 (linear)
#pragma unroll
  for (int k = 0; k < 4; ++k) a[k] = a_frag_bf16(s, DIM, k * 32, lane);
#pragma unroll
  for (int nt = 0; nt < 8; ++nt) {
    v8f acc = {};
#pragma unroll
    for (int k = 0; k < 4; ++k)
      acc = wmma_bf16(a[k], b_frag_wT(W3, nt * 16, k * 32, lane), acc);
    const float bias = b3[nt * 16 + cl];
#pragma unroll
    for (int v = 0; v < 8; ++v)
      y[(size_t)(mbase + v + 8 * lh) * DIM + nt * 16 + cl] = acc[v] + bias;
  }
}

// ---------------------------------------------------------------------------
// Fused LSTM cell step, in-place h/c update.
// gates = x@wih^T + h@whh^T + (bih+bhh);  i,f,g,o in 128-col blocks of N=512.
// One wave per 16 rows: 256 WMMAs per wave.
// ---------------------------------------------------------------------------
__global__ __launch_bounds__(128) void lstm_kernel(
    float* __restrict__ h, float* __restrict__ c, const float* __restrict__ x,
    const __bf16* __restrict__ wih, const __bf16* __restrict__ whh,
    const float* __restrict__ bsum) {
  const int lane = threadIdx.x & 31;
  const int wave = threadIdx.x >> 5;
  const int lh = lane >> 4;
  const int cl = lane & 15;
  const int mbase = blockIdx.x * 64 + wave * 16;

  v16bf ax[4], ah[4];
#pragma unroll
  for (int k = 0; k < 4; ++k) {
    ax[k] = a_frag_f32(x + (size_t)mbase * DIM, DIM, k * 32, lane);
    ah[k] = a_frag_f32(h + (size_t)mbase * DIM, DIM, k * 32, lane);
  }

  for (int t = 0; t < 8; ++t) {
    v8f gi = {}, gf = {}, gg = {}, go = {};
#pragma unroll
    for (int k = 0; k < 4; ++k) {
      gi = wmma_bf16(ax[k], b_frag_wT(wih, 0 * DIM + t * 16, k * 32, lane), gi);
      gi = wmma_bf16(ah[k], b_frag_wT(whh, 0 * DIM + t * 16, k * 32, lane), gi);
      gf = wmma_bf16(ax[k], b_frag_wT(wih, 1 * DIM + t * 16, k * 32, lane), gf);
      gf = wmma_bf16(ah[k], b_frag_wT(whh, 1 * DIM + t * 16, k * 32, lane), gf);
      gg = wmma_bf16(ax[k], b_frag_wT(wih, 2 * DIM + t * 16, k * 32, lane), gg);
      gg = wmma_bf16(ah[k], b_frag_wT(whh, 2 * DIM + t * 16, k * 32, lane), gg);
      go = wmma_bf16(ax[k], b_frag_wT(wih, 3 * DIM + t * 16, k * 32, lane), go);
      go = wmma_bf16(ah[k], b_frag_wT(whh, 3 * DIM + t * 16, k * 32, lane), go);
    }
    const int col = t * 16 + cl;
    const float bi = bsum[col];
    const float bff = bsum[DIM + col];
    const float bg = bsum[2 * DIM + col];
    const float bo = bsum[3 * DIM + col];
#pragma unroll
    for (int v = 0; v < 8; ++v) {
      const size_t idx = (size_t)(mbase + v + 8 * lh) * DIM + col;
      const float iv = sigmoidf(gi[v] + bi);
      const float fv = sigmoidf(gf[v] + bff);
      const float gv = tanhf(gg[v] + bg);
      const float ov = sigmoidf(go[v] + bo);
      const float c2 = fv * c[idx] + iv * gv;
      c[idx] = c2;
      h[idx] = ov * tanhf(c2);
    }
  }
}

// ---------------------------------------------------------------------------
// Vote head: relu(relu(x W1^T+b1) W2^T+b2) . w3 + b3  (final dim = 1)
// ---------------------------------------------------------------------------
__global__ __launch_bounds__(128) void vote_kernel(
    const float* __restrict__ x, const __bf16* __restrict__ W1,
    const float* __restrict__ b1, const __bf16* __restrict__ W2,
    const float* __restrict__ b2, const float* __restrict__ w3,
    const float* __restrict__ b3, float* __restrict__ out) {
  __shared__ __bf16 sb[4][16 * DIM];
  __shared__ float sf[4][16 * DIM];
  const int lane = threadIdx.x & 31;
  const int wave = threadIdx.x >> 5;
  const int lh = lane >> 4;
  const int cl = lane & 15;
  const int mbase = blockIdx.x * 64 + wave * 16;
  __bf16* s = sb[wave];
  float* sfl = sf[wave];

  v16bf a[4];
#pragma unroll
  for (int k = 0; k < 4; ++k)
    a[k] = a_frag_f32(x + (size_t)mbase * DIM, DIM, k * 32, lane);
#pragma unroll
  for (int nt = 0; nt < 8; ++nt) {
    v8f acc = {};
#pragma unroll
    for (int k = 0; k < 4; ++k)
      acc = wmma_bf16(a[k], b_frag_wT(W1, nt * 16, k * 32, lane), acc);
    const float bias = b1[nt * 16 + cl];
#pragma unroll
    for (int v = 0; v < 8; ++v) {
      float t = acc[v] + bias;
      s[(v + 8 * lh) * DIM + nt * 16 + cl] = (__bf16)fmaxf(t, 0.0f);
    }
  }
#pragma unroll
  for (int k = 0; k < 4; ++k) a[k] = a_frag_bf16(s, DIM, k * 32, lane);
#pragma unroll
  for (int nt = 0; nt < 8; ++nt) {
    v8f acc = {};
#pragma unroll
    for (int k = 0; k < 4; ++k)
      acc = wmma_bf16(a[k], b_frag_wT(W2, nt * 16, k * 32, lane), acc);
    const float bias = b2[nt * 16 + cl];
#pragma unroll
    for (int v = 0; v < 8; ++v) {
      float t = acc[v] + bias;
      sfl[(v + 8 * lh) * DIM + nt * 16 + cl] = fmaxf(t, 0.0f);
    }
  }
  __syncthreads();
  if (lane < 16) {
    const float* row = sfl + lane * DIM;
    float acc = b3[0];
#pragma unroll 8
    for (int k = 0; k < DIM; ++k) acc += row[k] * w3[k];
    out[mbase + lane] = acc;  // mean over axis of size 1 == value
  }
}

// ---------------------------------------------------------------------------
// SpMM via L2 atomics: dst[didx[e]] += src[sidx[e]]  (rows of DIM floats).
// One wave per edge, lane covers 4 contiguous floats.
// ---------------------------------------------------------------------------
__global__ __launch_bounds__(256) void scatter_add_kernel(
    const int* __restrict__ sidx, const int* __restrict__ didx,
    const float* __restrict__ src, float* __restrict__ dst, int nedges) {
  const int t = blockIdx.x * blockDim.x + threadIdx.x;
  const int e = t >> 5;
  const int part = t & 31;
  if (e >= nedges) return;
  const int s = sidx[e];
  const int d = didx[e];
  const float4 v = *(const float4*)(src + (size_t)s * DIM + part * 4);
  float* dp = dst + (size_t)d * DIM + part * 4;
  atomicAdd(dp + 0, v.x);
  atomicAdd(dp + 1, v.y);
  atomicAdd(dp + 2, v.z);
  atomicAdd(dp + 3, v.w);
}

// ---------------------------------------------------------------------------
// Small setup kernels
// ---------------------------------------------------------------------------
__global__ void cvt_bf16_kernel(const float* __restrict__ in,
                                __bf16* __restrict__ out, int n) {
  int i = blockIdx.x * blockDim.x + threadIdx.x;
  if (i < n) out[i] = (__bf16)in[i];
}

__global__ void add_kernel(const float* __restrict__ a, const float* __restrict__ b,
                           float* __restrict__ o, int n) {
  int i = blockIdx.x * blockDim.x + threadIdx.x;
  if (i < n) o[i] = a[i] + b[i];
}

__global__ void init_state_kernel(const float* __restrict__ w,
                                  const float* __restrict__ b,
                                  float* __restrict__ h, float* __restrict__ c,
                                  int total) {
  int i = blockIdx.x * blockDim.x + threadIdx.x;
  if (i < total) {
    int d = i & (DIM - 1);
    h[i] = w[d] + b[d];  // ones(1) @ w.T + b, w is [DIM,1]
    c[i] = 0.0f;
  }
}

__global__ void zero_kernel(float* __restrict__ p, int n) {
  int i = blockIdx.x * blockDim.x + threadIdx.x;
  if (i < n) p[i] = 0.0f;
}

// ---------------------------------------------------------------------------
// Host orchestration
// ---------------------------------------------------------------------------
// d_in flattening (setup_inputs dict order, depth-first):
//  0 unpack_rows  1 unpack_cols
//  2 var_init.w   3 var_init.b   4 node_init.w  5 node_init.b
//  6 var_msg.l1.w 7 .l1.b  8 .l2.w  9 .l2.b 10 .l3.w 11 .l3.b
// 12 node_msg.l1.w ... 17 node_msg.l3.b
// 18 var_update.wih 19 .whh 20 .bih 21 .bhh
// 22 node_update.wih 23 .whh 24 .bih 25 .bhh
// 26 node_vote.l1.w 27 .l1.b 28 .l2.w 29 .l2.b 30 .l3.w 31 .l3.b

extern "C" void kernel_launch(void* const* d_in, const int* in_sizes, int n_in,
                              void* d_out, int out_size, void* d_ws, size_t ws_size,
                              hipStream_t stream) {
  (void)in_sizes; (void)n_in; (void)out_size; (void)ws_size;
  const int* rows = (const int*)d_in[0];
  const int* cols = (const int*)d_in[1];
  auto F = [&](int i) { return (const float*)d_in[i]; };

  char* base = (char*)d_ws;
  size_t off = 0;
  auto alloc = [&](size_t bytes) -> void* {
    void* p = base + off;
    off += (bytes + 255) & ~(size_t)255;
    return p;
  };

  float* var_h    = (float*)alloc((size_t)N_VAR * DIM * 4);
  float* var_c    = (float*)alloc((size_t)N_VAR * DIM * 4);
  float* var_pre  = (float*)alloc((size_t)N_VAR * DIM * 4);
  float* p2c      = (float*)alloc((size_t)N_VAR * DIM * 4);
  float* node_h   = (float*)alloc((size_t)N_NODE * DIM * 4);
  float* node_c   = (float*)alloc((size_t)N_NODE * DIM * 4);
  float* node_pre = (float*)alloc((size_t)N_NODE * DIM * 4);
  float* c2p      = (float*)alloc((size_t)N_NODE * DIM * 4);

  __bf16* w_vmsg1 = (__bf16*)alloc((size_t)DIM * DIM * 2);
  __bf16* w_vmsg2 = (__bf16*)alloc((size_t)DIM * DIM * 2);
  __bf16* w_vmsg3 = (__bf16*)alloc((size_t)DIM * DIM * 2);
  __bf16* w_nmsg1 = (__bf16*)alloc((size_t)DIM * DIM * 2);
  __bf16* w_nmsg2 = (__bf16*)alloc((size_t)DIM * DIM * 2);
  __bf16* w_nmsg3 = (__bf16*)alloc((size_t)DIM * DIM * 2);
  __bf16* w_vote1 = (__bf16*)alloc((size_t)DIM * DIM * 2);
  __bf16* w_vote2 = (__bf16*)alloc((size_t)DIM * DIM * 2);
  __bf16* w_vih   = (__bf16*)alloc((size_t)4 * DIM * DIM * 2);
  __bf16* w_vhh   = (__bf16*)alloc((size_t)4 * DIM * DIM * 2);
  __bf16* w_nih   = (__bf16*)alloc((size_t)4 * DIM * DIM * 2);
  __bf16* w_nhh   = (__bf16*)alloc((size_t)4 * DIM * DIM * 2);
  float* bsum_v   = (float*)alloc(4 * DIM * 4);
  float* bsum_n   = (float*)alloc(4 * DIM * 4);

  auto CVT = [&](int idx, __bf16* dst, int n) {
    cvt_bf16_kernel<<<(n + 255) / 256, 256, 0, stream>>>(F(idx), dst, n);
  };
  CVT(6, w_vmsg1, DIM * DIM);
  CVT(8, w_vmsg2, DIM * DIM);
  CVT(10, w_vmsg3, DIM * DIM);
  CVT(12, w_nmsg1, DIM * DIM);
  CVT(14, w_nmsg2, DIM * DIM);
  CVT(16, w_nmsg3, DIM * DIM);
  CVT(18, w_vih, 4 * DIM * DIM);
  CVT(19, w_vhh, 4 * DIM * DIM);
  CVT(22, w_nih, 4 * DIM * DIM);
  CVT(23, w_nhh, 4 * DIM * DIM);
  CVT(26, w_vote1, DIM * DIM);
  CVT(28, w_vote2, DIM * DIM);
  add_kernel<<<2, 256, 0, stream>>>(F(20), F(21), bsum_v, 4 * DIM);
  add_kernel<<<2, 256, 0, stream>>>(F(24), F(25), bsum_n, 4 * DIM);

  init_state_kernel<<<(N_VAR * DIM) / 256, 256, 0, stream>>>(F(2), F(3), var_h,
                                                             var_c, N_VAR * DIM);
  init_state_kernel<<<(N_NODE * DIM) / 256, 256, 0, stream>>>(F(4), F(5), node_h,
                                                              node_c, N_NODE * DIM);

  const int scatter_blocks = (N_CELLS * 32) / 256;
  for (int r = 0; r < N_ROUNDS; ++r) {
    // var_pre = MLP(var_msg, var_h)
    mlp3_kernel<<<N_VAR / 64, 128, 0, stream>>>(var_h, var_pre, w_vmsg1, F(7),
                                                w_vmsg2, F(9), w_vmsg3, F(11));
    // c2p = unpack^T @ var_pre
    zero_kernel<<<(N_NODE * DIM) / 256, 256, 0, stream>>>(c2p, N_NODE * DIM);
    scatter_add_kernel<<<scatter_blocks, 256, 0, stream>>>(rows, cols, var_pre,
                                                           c2p, N_CELLS);
    // node state update (reference uses var_update params here)
    lstm_kernel<<<N_NODE / 64, 128, 0, stream>>>(node_h, node_c, c2p, w_vih,
                                                 w_vhh, bsum_v);
    // node_pre = MLP(node_msg, node_h)
    mlp3_kernel<<<N_NODE / 64, 128, 0, stream>>>(node_h, node_pre, w_nmsg1, F(13),
                                                 w_nmsg2, F(15), w_nmsg3, F(17));
    // p2c = unpack @ node_pre
    zero_kernel<<<(N_VAR * DIM) / 256, 256, 0, stream>>>(p2c, N_VAR * DIM);
    scatter_add_kernel<<<scatter_blocks, 256, 0, stream>>>(cols, rows, node_pre,
                                                           p2c, N_CELLS);
    // var state update (reference uses node_update params here)
    lstm_kernel<<<N_VAR / 64, 128, 0, stream>>>(var_h, var_c, p2c, w_nih, w_nhh,
                                                bsum_n);
  }

  vote_kernel<<<N_VAR / 64, 128, 0, stream>>>(var_h, w_vote1, F(27), w_vote2,
                                              F(29), F(30), F(31), (float*)d_out);
}
